// WWW19Layer_21818433864127
// MI455X (gfx1250) — matrix operand order repaired and checked by reading, not compiled, not run on hardware
//
#include <hip/hip_runtime.h>

#define BB 16
#define LM 1024
#define LX 1024
#define DD 768

typedef __attribute__((ext_vector_type(16))) __bf16 v16bf;
typedef __attribute__((ext_vector_type(8)))  float  v8f;

// ---------------------------------------------------------------------------
// Vectorized WMMA fragment helpers (CDNA5 wave32 layouts, CDNA5 ISA §7.12.2)
// ---------------------------------------------------------------------------

// A fragment 16x32: lane holds row m0+(lane&15); halfs = p[0..15]
// where p = row + k0 + 16*(lane>=16).  Two aligned b128 loads.
__device__ inline v16bf load_a_frag(const __bf16* base, int ld, int m0, int k0, int lane) {
    const int r    = m0 + (lane & 15);
    const int kadd = (lane >= 16) ? 16 : 0;
    const __bf16* p = base + (size_t)r * ld + (k0 + kadd);
    union { uint4 q[2]; v16bf v; } u;
    u.q[0] = ((const uint4*)p)[0];   // halfs 0..7   (K = k0+kadd+0..7)
    u.q[1] = ((const uint4*)p)[1];   // halfs 8..15  (K = k0+kadd+8..15)
    return u.v;
}

// B fragment 32x16 from a TRANSPOSED (N-major) source BT[n*ldT + k]:
// lane holds column n0+(lane&15); halfs i -> K = k0 + 16*(lane>=16) + i,
// i.e. 16 contiguous halfs.  Two aligned b128 loads.
__device__ inline v16bf load_b_fragT(const __bf16* baseT, int ldT, int k0, int n0, int lane) {
    const int n  = n0 + (lane & 15);
    const int kb = k0 + ((lane >= 16) ? 16 : 0);
    const __bf16* p = baseT + (size_t)n * ldT + kb;
    union { uint4 q[2]; v16bf v; } u;
    u.q[0] = ((const uint4*)p)[0];
    u.q[1] = ((const uint4*)p)[1];
    return u.v;
}

// C/D 16x16 f32: reg r of lane L -> M = m0 + r + 8*(L>=16), N = n0 + (L&15).
// Transposed bf16 store: per lane the 8 rows are CONTIGUOUS -> one b128 store.
__device__ inline void store_tile_bf16_T(__bf16* CT, int ldT, int m0, int n0, int lane, v8f acc) {
    const int n  = n0 + (lane & 15);
    const int mb = m0 + ((lane >= 16) ? 8 : 0);
    union { uint4 q; __bf16 h[8]; } u;
#pragma unroll
    for (int r = 0; r < 8; ++r) u.h[r] = (__bf16)acc[r];
    *(uint4*)(CT + (size_t)n * ldT + mb) = u.q;
}

// Row-major f32 tile store (strided, coalesced across lanes)
__device__ inline void store_tile_f32(float* C, int ld, int m0, int n0, int lane, v8f acc) {
    const int n  = n0 + (lane & 15);
    const int mb = m0 + ((lane >= 16) ? 8 : 0);
#pragma unroll
    for (int r = 0; r < 8; ++r)
        C[(size_t)(mb + r) * ld + n] = acc[r];
}

// ---------------------------------------------------------------------------
// Double-buffered 16x64 GEMM strip: D[16x64] += A[16xK] * BT[64xK]^T.
// Loads for step k+32 are issued BEFORE the WMMAs of step k so the WMMAs only
// depend on the previous iteration's loads (partial s_wait_loadcnt, VMEM/WMMA
// overlap).  Branch-free clamped next-k keeps loads unconditional.
// ---------------------------------------------------------------------------
__device__ inline void gemm_strip(const __bf16* __restrict__ Abase, int lda,
                                  const __bf16* __restrict__ BTbase, int ldb,
                                  int K, int m0, int n0, int lane,
                                  v8f& acc0, v8f& acc1, v8f& acc2, v8f& acc3) {
    v16bf a  = load_a_frag (Abase,  lda, m0, 0, lane);
    v16bf b0 = load_b_fragT(BTbase, ldb, 0, n0 +  0, lane);
    v16bf b1 = load_b_fragT(BTbase, ldb, 0, n0 + 16, lane);
    v16bf b2 = load_b_fragT(BTbase, ldb, 0, n0 + 32, lane);
    v16bf b3 = load_b_fragT(BTbase, ldb, 0, n0 + 48, lane);
    for (int k = 0; k < K; k += 32) {
        const int kn = (k + 32 < K) ? (k + 32) : k;   // clamped, branch-free
        v16bf an  = load_a_frag (Abase,  lda, m0, kn, lane);
        v16bf bn0 = load_b_fragT(BTbase, ldb, kn, n0 +  0, lane);
        v16bf bn1 = load_b_fragT(BTbase, ldb, kn, n0 + 16, lane);
        v16bf bn2 = load_b_fragT(BTbase, ldb, kn, n0 + 32, lane);
        v16bf bn3 = load_b_fragT(BTbase, ldb, kn, n0 + 48, lane);
        acc0 = __builtin_amdgcn_wmma_f32_16x16x32_bf16(false, a, false, b0, (short)0, acc0, false, false);
        acc1 = __builtin_amdgcn_wmma_f32_16x16x32_bf16(false, a, false, b1, (short)0, acc1, false, false);
        acc2 = __builtin_amdgcn_wmma_f32_16x16x32_bf16(false, a, false, b2, (short)0, acc2, false, false);
        acc3 = __builtin_amdgcn_wmma_f32_16x16x32_bf16(false, a, false, b3, (short)0, acc3, false, false);
        a = an; b0 = bn0; b1 = bn1; b2 = bn2; b3 = bn3;
    }
}

// ---------------------------------------------------------------------------
// Kernel 0a: f32 -> bf16 conversion (grid-stride)
// ---------------------------------------------------------------------------
__global__ void f32_to_bf16_kernel(const float* __restrict__ in,
                                   __bf16* __restrict__ out, long n) {
    long i = (long)blockIdx.x * blockDim.x + threadIdx.x;
    const long stride = (long)gridDim.x * blockDim.x;
    for (; i < n; i += stride) out[i] = (__bf16)in[i];
}

// ---------------------------------------------------------------------------
// Kernel 0b: x (B,LX,D) f32 -> xT (B,D,LX) bf16, LDS-tiled 32x32 transpose
// ---------------------------------------------------------------------------
__global__ void transpose_bf16_kernel(const float* __restrict__ x,
                                      __bf16* __restrict__ xT) {
    __shared__ float t[32][33];
    const int b  = blockIdx.z;
    const int l0 = blockIdx.y * 32;
    const int d0 = blockIdx.x * 32;
    const int tx = threadIdx.x & 31;
    const int ty = threadIdx.x >> 5;          // 0..7

    const float* xb = x + (size_t)b * LX * DD;
#pragma unroll
    for (int i = 0; i < 32; i += 8)
        t[ty + i][tx] = xb[(size_t)(l0 + ty + i) * DD + (d0 + tx)];
    __syncthreads();

    __bf16* xtb = xT + (size_t)b * DD * LX;
#pragma unroll
    for (int i = 0; i < 32; i += 8)
        xtb[(size_t)(d0 + ty + i) * LX + (l0 + tx)] = (__bf16)t[tx][ty + i];
}

// ---------------------------------------------------------------------------
// Kernel 1: G_b = x_b^T @ x_b  (768x768, K=1024).  A and B both from xT.
// Output stored "transposed" == row-major (G symmetric), packed b128 stores.
// ---------------------------------------------------------------------------
__global__ void gram_kernel(const __bf16* __restrict__ xT, __bf16* __restrict__ G) {
    const int lane = threadIdx.x & 31;
    const int tile = blockIdx.x * 8 + (threadIdx.x >> 5);
    const int NT = DD / 64;                  // 12
    const int MT = DD / 16;                  // 48
    const int nt = tile % NT;
    const int mt = (tile / NT) % MT;
    const int b  = tile / (NT * MT);

    const __bf16* xt = xT + (size_t)b * DD * LX;
    __bf16*       g  = G  + (size_t)b * DD * DD;
    const int m0 = mt * 16, n0 = nt * 64;

    v8f acc0 = {}, acc1 = {}, acc2 = {}, acc3 = {};
    gemm_strip(xt, LX, xt, LX, LX, m0, n0, lane, acc0, acc1, acc2, acc3);

    store_tile_bf16_T(g, DD, m0, n0 +  0, lane, acc0);
    store_tile_bf16_T(g, DD, m0, n0 + 16, lane, acc1);
    store_tile_bf16_T(g, DD, m0, n0 + 32, lane, acc2);
    store_tile_bf16_T(g, DD, m0, n0 + 48, lane, acc3);
}

// ---------------------------------------------------------------------------
// Kernel 2: M_b = W @ G_b (768x768, K=768).  B-frags read G[n*768+k] == G[k][n]
// by symmetry.  Output stored TRANSPOSED (MT) for kernel 3.
// ---------------------------------------------------------------------------
__global__ void wg_kernel(const __bf16* __restrict__ Wb, const __bf16* __restrict__ G,
                          __bf16* __restrict__ MT) {
    const int lane = threadIdx.x & 31;
    const int tile = blockIdx.x * 8 + (threadIdx.x >> 5);
    const int NT = DD / 64;                  // 12
    const int MTt = DD / 16;                 // 48
    const int nt = tile % NT;
    const int mt = (tile / NT) % MTt;
    const int b  = tile / (NT * MTt);

    const __bf16* g  = G  + (size_t)b * DD * DD;
    __bf16*       mo = MT + (size_t)b * DD * DD;
    const int m0 = mt * 16, n0 = nt * 64;

    v8f acc0 = {}, acc1 = {}, acc2 = {}, acc3 = {};
    gemm_strip(Wb, DD, g, DD, DD, m0, n0, lane, acc0, acc1, acc2, acc3);

    store_tile_bf16_T(mo, DD, m0, n0 +  0, lane, acc0);
    store_tile_bf16_T(mo, DD, m0, n0 + 16, lane, acc1);
    store_tile_bf16_T(mo, DD, m0, n0 + 32, lane, acc2);
    store_tile_bf16_T(mo, DD, m0, n0 + 48, lane, acc3);
}

// ---------------------------------------------------------------------------
// Kernel 3: A_b = main_b @ M_b (1024x768, K=768), f32 output row-major.
// ---------------------------------------------------------------------------
__global__ void attend_kernel(const __bf16* __restrict__ mainb,
                              const __bf16* __restrict__ MT,
                              float* __restrict__ A) {
    const int lane = threadIdx.x & 31;
    const int tile = blockIdx.x * 8 + (threadIdx.x >> 5);
    const int NT = DD / 64;                  // 12
    const int MTt = LM / 16;                 // 64
    const int nt = tile % NT;
    const int mt = (tile / NT) % MTt;
    const int b  = tile / (NT * MTt);

    const __bf16* mm = mainb + (size_t)b * LM * DD;
    const __bf16* mb = MT    + (size_t)b * DD * DD;
    float*        ab = A     + (size_t)b * LM * DD;
    const int m0 = mt * 16, n0 = nt * 64;

    v8f acc0 = {}, acc1 = {}, acc2 = {}, acc3 = {};
    gemm_strip(mm, DD, mb, DD, DD, m0, n0, lane, acc0, acc1, acc2, acc3);

    store_tile_f32(ab, DD, m0, n0 +  0, lane, acc0);
    store_tile_f32(ab, DD, m0, n0 + 16, lane, acc1);
    store_tile_f32(ab, DD, m0, n0 + 32, lane, acc2);
    store_tile_f32(ab, DD, m0, n0 + 48, lane, acc3);
}

// ---------------------------------------------------------------------------
// Kernel 4: beta[b,m] — one wave per row, fixed shuffle tree (deterministic)
// ---------------------------------------------------------------------------
__global__ void beta_kernel(const float* __restrict__ mainf,
                            const float* __restrict__ Af,
                            const float* __restrict__ w,
                            float* __restrict__ beta) {
    const int lane = threadIdx.x & 31;
    const int row  = blockIdx.x * 8 + (threadIdx.x >> 5);
    const float* mr = mainf + (size_t)row * DD;
    const float* ar = Af    + (size_t)row * DD;
    float p = 0.0f;
    for (int k = lane; k < DD; k += 32) {
        const float mv = mr[k], av = ar[k];
        p += (mv - av) * w[k] + (mv * av) * w[DD + k];
    }
#pragma unroll
    for (int off = 16; off > 0; off >>= 1) p += __shfl_xor(p, off, 32);
    if (lane == 0) beta[row] = p;
}

// ---------------------------------------------------------------------------
// Kernel 5: pooled[b,k] = sum_m beta[b,m] * out(b,m,k)
// beta staged into LDS with gfx1250 async global->LDS (ASYNCcnt path).
// ---------------------------------------------------------------------------
__global__ void pool_kernel(const float* __restrict__ mainf,
                            const float* __restrict__ Af,
                            const float* __restrict__ beta,
                            float* __restrict__ out) {
    __shared__ float sb[LM];
    const int chunks = (2 * DD) / 256;      // 6
    const int b     = blockIdx.x / chunks;
    const int chunk = blockIdx.x % chunks;
    const int tid   = threadIdx.x;

    // async copy: 256 threads x 16B = 4KB = all of beta[b, 0..1023]
    {
        unsigned int  ldsoff = (unsigned int)(uintptr_t)(&sb[tid * 4]);
        const float*  g      = beta + (size_t)b * LM + tid * 4;
        asm volatile("global_load_async_to_lds_b128 %0, %1, off"
                     :: "v"(ldsoff), "v"(g) : "memory");
        asm volatile("s_wait_asynccnt 0x0" ::: "memory");
    }
    __syncthreads();

    const int k = chunk * 256 + tid;        // [0, 1536)
    const float* mb = mainf + (size_t)b * LM * DD;
    const float* ab = Af    + (size_t)b * LM * DD;
    float acc = 0.0f;
    if (k < DD) {
        for (int m = 0; m < LM; ++m) {
            const float mv = mb[(size_t)m * DD + k], av = ab[(size_t)m * DD + k];
            acc += sb[m] * (mv - av);
        }
    } else {
        const int kk = k - DD;
        for (int m = 0; m < LM; ++m) {
            const float mv = mb[(size_t)m * DD + kk], av = ab[(size_t)m * DD + kk];
            acc += sb[m] * (mv * av);
        }
    }
    out[(size_t)b * 2 * DD + k] = acc;
}

// ---------------------------------------------------------------------------
// Launch
// ---------------------------------------------------------------------------
extern "C" void kernel_launch(void* const* d_in, const int* in_sizes, int n_in,
                              void* d_out, int out_size, void* d_ws, size_t ws_size,
                              hipStream_t stream) {
    (void)in_sizes; (void)n_in; (void)out_size; (void)ws_size;

    const float* mainF = (const float*)d_in[0];   // (B, LM, D)
    const float* xF    = (const float*)d_in[1];   // (B, LX, D)
    const float* WF    = (const float*)d_in[2];   // (D, D)
    const float* wF    = (const float*)d_in[3];   // (2D, 1)
    float* out = (float*)d_out;

    char* ws = (char*)d_ws;
    const size_t xt_off   = 0;                                   // B*D*LX*2
    const size_t mbf_off  = xt_off   + (size_t)BB * DD * LX * 2; // B*LM*D*2
    const size_t wbf_off  = mbf_off  + (size_t)BB * LM * DD * 2; // D*D*2
    const size_t g_off    = wbf_off  + (size_t)DD * DD * 2;      // B*D*D*2
    const size_t mt_off   = g_off    + (size_t)BB * DD * DD * 2; // B*D*D*2
    const size_t a_off    = mt_off   + (size_t)BB * DD * DD * 2; // B*LM*D*4
    const size_t beta_off = a_off    + (size_t)BB * LM * DD * 4; // B*LM*4

    __bf16* xT   = (__bf16*)(ws + xt_off);
    __bf16* mbf  = (__bf16*)(ws + mbf_off);
    __bf16* wbf  = (__bf16*)(ws + wbf_off);
    __bf16* G    = (__bf16*)(ws + g_off);
    __bf16* MT   = (__bf16*)(ws + mt_off);
    float*  Af   = (float*)(ws + a_off);
    float*  beta = (float*)(ws + beta_off);

    // 0) conversions
    transpose_bf16_kernel<<<dim3(DD / 32, LX / 32, BB), 256, 0, stream>>>(xF, xT);
    f32_to_bf16_kernel<<<2048, 256, 0, stream>>>(mainF, mbf, (long)BB * LM * DD);
    f32_to_bf16_kernel<<<512,  256, 0, stream>>>(WF,    wbf, (long)DD * DD);

    // 1) G_b = x_b^T x_b : B*48*12 wave-strips, 8 waves/block
    gram_kernel<<<(BB * 48 * 12) / 8, 256, 0, stream>>>(xT, G);

    // 2) M_b = W @ G_b (stored transposed)
    wg_kernel<<<(BB * 48 * 12) / 8, 256, 0, stream>>>(wbf, G, MT);

    // 3) A_b = main_b @ M_b
    attend_kernel<<<(BB * 64 * 12) / 8, 256, 0, stream>>>(mbf, MT, Af);

    // 4) beta
    beta_kernel<<<(BB * LM) / 8, 256, 0, stream>>>(mainF, Af, wF, beta);

    // 5) pooled
    pool_kernel<<<BB * 6, 256, 0, stream>>>(mainF, Af, beta, out);
}
